// AttnDecoderRNN_78013785965118
// MI455X (gfx1250) — compile-verified, matrix operand order checked by reading, not run on hardware
//
#include <hip/hip_runtime.h>

typedef __bf16 bf16;
typedef __attribute__((ext_vector_type(8)))  bf16  v8bf;
typedef __attribute__((ext_vector_type(16))) bf16  v16bf;
typedef __attribute__((ext_vector_type(8)))  float v8f;

#define B_  32
#define S_  64
#define T_  64
#define H_  1024
#define V_  32000

// ---------------------------------------------------------------------------
// helpers
// ---------------------------------------------------------------------------
__device__ inline v16bf pack16(v8bf lo, v8bf hi) {
  v16bf r;
#pragma unroll
  for (int i = 0; i < 8; ++i) { r[i] = lo[i]; r[8 + i] = hi[i]; }
  return r;
}

// Stage a 32xK bf16 A-tile into LDS using CDNA5 async copies (ASYNCcnt path,
// no VGPR round-trip), then wait + barrier.
__device__ inline void stage_A_async(bf16* lA, const bf16* Ab, int K) {
  const int chunks = (32 * K) / 8;  // 16-byte chunks
  for (int i = threadIdx.x; i < chunks; i += 256) {
    unsigned      lo = (unsigned)(uintptr_t)(lA + (size_t)i * 8);
    unsigned long ga = (unsigned long)(uintptr_t)(Ab + (size_t)i * 8);
    asm volatile("global_load_async_to_lds_b128 %0, %1, off"
                 :
                 : "v"(lo), "v"(ga)
                 : "memory");
  }
  asm volatile("s_wait_asynccnt 0" ::: "memory");
  __syncthreads();
}

// Core 32xN WMMA pass: one wave owns a 32(M) x 16(N) tile; per K-step two
// v_wmma_f32_16x16x32_bf16 reuse the B fragment across both M accumulators.
__device__ inline void wmma_pass(const bf16* lA, const bf16* W,
                                 const float* bias, float* C, int K, long ldc,
                                 long mbase, int nbase, int nt) {
  const int wave = threadIdx.x >> 5;
  const int lane = threadIdx.x & 31;
  const int col  = nbase + wave * 16 + (lane & 15);
  const int koff = (lane & 16) ? 8 : 0;  // per-half-wave K sub-pattern

  const bf16* wr = W  + (size_t)col * K + koff;          // B: lane = column
  const bf16* a0 = lA + (size_t)(lane & 15) * K + koff;  // A rows 0..15
  const bf16* a1 = a0 + 16 * (size_t)K;                  // A rows 16..31

  v8f c0 = {}; v8f c1 = {};
  for (int kb = 0; kb < K; kb += 32) {
    v16bf bfrag = pack16(*(const v8bf*)(wr + kb), *(const v8bf*)(wr + kb + 16));
    v16bf af0   = pack16(*(const v8bf*)(a0 + kb), *(const v8bf*)(a0 + kb + 16));
    v16bf af1   = pack16(*(const v8bf*)(a1 + kb), *(const v8bf*)(a1 + kb + 16));
    c0 = __builtin_amdgcn_wmma_f32_16x16x32_bf16(false, af0, false, bfrag,
                                                 (short)0, c0, false, false);
    c1 = __builtin_amdgcn_wmma_f32_16x16x32_bf16(false, af1, false, bfrag,
                                                 (short)0, c1, false, false);
  }

  const float bv = bias ? bias[col] : 0.0f;
  const int m0 = (lane & 16) ? 8 : 0;  // D layout: lanes 16-31 hold M=8..15
  float* Cb = C + mbase * ldc + col;
  if (nt) {  // streaming output (262MB logits) -> don't thrash L2-resident W
#pragma unroll
    for (int r = 0; r < 8; ++r) {
      __builtin_nontemporal_store(c0[r] + bv, Cb + (size_t)(m0 + r) * ldc);
      __builtin_nontemporal_store(c1[r] + bv, Cb + (size_t)(m0 + r + 16) * ldc);
    }
  } else {
#pragma unroll
    for (int r = 0; r < 8; ++r) {
      Cb[(size_t)(m0 + r)      * ldc] = c0[r] + bv;
      Cb[(size_t)(m0 + r + 16) * ldc] = c1[r] + bv;
    }
  }
}

// ---------------------------------------------------------------------------
// f32 -> bf16 conversion (weights converted once per launch; resident in L2)
// ---------------------------------------------------------------------------
__global__ __launch_bounds__(256) void f32_to_bf16_kernel(
    const float* __restrict__ src, bf16* __restrict__ dst, long n) {
  long i = (long)blockIdx.x * 256 + threadIdx.x;
  if (i < n) dst[i] = (bf16)src[i];
}

__global__ __launch_bounds__(256) void init_hidden_kernel(
    const float* __restrict__ eh, float* __restrict__ hidf,
    bf16* __restrict__ hidb) {
  int i = blockIdx.x * 256 + threadIdx.x;
  if (i < B_ * H_) {
    float v = eh[i];
    hidf[i] = v;
    hidb[i] = (bf16)v;
  }
}

// ---------------------------------------------------------------------------
// Generic bf16 WMMA GEMM:  C[m, n] = sum_k A[m,k] * W[n,k]  (+ bias[n])
// block = 256 threads = 8 wave32 covering 128 N; gridDim.y tiles M by 32.
// ---------------------------------------------------------------------------
__global__ __launch_bounds__(256) void gemm_bf16_wmma_kernel(
    const bf16* __restrict__ A, const bf16* __restrict__ W,
    const float* __restrict__ bias, float* __restrict__ C,
    int K, long ldc, int nt) {
  extern __shared__ bf16 lA[];  // 32 * K bf16
  stage_A_async(lA, A + (size_t)blockIdx.y * 32 * K, K);
  wmma_pass(lA, W, bias, C, K, ldc, (long)blockIdx.y * 32,
            blockIdx.x * 128, nt);
}

// Fused GRU-gate GEMMs (saves one graph node per step):
//   z==0: gi = x2  @ W_ih.T + b_ih   (K=2H)
//   z==1: gh = hid @ W_hh.T + b_hh   (K=H)
__global__ __launch_bounds__(256) void gemm_gates_kernel(
    const bf16* __restrict__ A0, const bf16* __restrict__ W0,
    const float* __restrict__ b0, float* __restrict__ C0, int K0,
    const bf16* __restrict__ A1, const bf16* __restrict__ W1,
    const float* __restrict__ b1, float* __restrict__ C1, int K1,
    long ldc) {
  extern __shared__ bf16 lA[];
  const bf16* A; const bf16* W; const float* bias; float* C; int K;
  if (blockIdx.z == 0) { A = A0; W = W0; bias = b0; C = C0; K = K0; }
  else                 { A = A1; W = W1; bias = b1; C = C1; K = K1; }
  stage_A_async(lA, A, K);
  wmma_pass(lA, W, bias, C, K, ldc, 0, blockIdx.x * 128, 0);
}

// ---------------------------------------------------------------------------
// Attention: scores = Va . tanh(q + Uk), softmax over S, ctx = w @ enc.
// Also writes attention output and builds xi = [emb[tok], ctx] in bf16.
// One block per batch row.
// ---------------------------------------------------------------------------
__global__ __launch_bounds__(256) void attn_ctx_kernel(
    const float* __restrict__ q,       // [B,H]
    const float* __restrict__ Uk,      // [B,S,H]
    const float* __restrict__ Va_w,    // [H]
    const float* __restrict__ Va_b,    // [1]
    const float* __restrict__ enc,     // [B,S,H]
    const int*   __restrict__ target,  // [B,T]
    const float* __restrict__ emb,     // [V,H]
    float* __restrict__ ctx,           // [B,H]
    bf16*  __restrict__ x2,            // [B,2H]
    float* __restrict__ attn_base,     // [B,T,S]
    int t) {
  __shared__ float sc[S_];
  __shared__ float wts[S_];
  const int b = blockIdx.x;
  const int wave = threadIdx.x >> 5;
  const int lane = threadIdx.x & 31;
  const float* qb = q + (size_t)b * H_;

  for (int s = wave; s < S_; s += 8) {
    const float* uk = Uk + ((size_t)b * S_ + s) * H_;
    float p = 0.0f;
    for (int h = lane; h < H_; h += 32)
      p += Va_w[h] * tanhf(qb[h] + uk[h]);
#pragma unroll
    for (int off = 16; off > 0; off >>= 1) p += __shfl_xor(p, off, 32);
    if (lane == 0) sc[s] = p + Va_b[0];
  }
  __syncthreads();

  if (threadIdx.x < 32) {  // softmax over 64 values by one wave32
    float v0 = sc[lane], v1 = sc[lane + 32];
    float m = fmaxf(v0, v1);
#pragma unroll
    for (int off = 16; off > 0; off >>= 1) m = fmaxf(m, __shfl_xor(m, off, 32));
    float e0 = expf(v0 - m), e1 = expf(v1 - m);
    float sum = e0 + e1;
#pragma unroll
    for (int off = 16; off > 0; off >>= 1) sum += __shfl_xor(sum, off, 32);
    float inv = 1.0f / sum;
    wts[lane] = e0 * inv;
    wts[lane + 32] = e1 * inv;
    float* ao = attn_base + (size_t)b * T_ * S_ + (size_t)t * S_;
    ao[lane] = e0 * inv;
    ao[lane + 32] = e1 * inv;
  }
  __syncthreads();

  const int tok = (t == 0) ? 1 : target[b * T_ + (t - 1)];
  const float* er = emb + (size_t)tok * H_;
  for (int h = threadIdx.x; h < H_; h += 256) {
    float acc = 0.0f;
    const float* eb = enc + (size_t)b * S_ * H_ + h;
#pragma unroll 4
    for (int s = 0; s < S_; ++s) acc += wts[s] * eb[(size_t)s * H_];
    ctx[(size_t)b * H_ + h] = acc;
    x2[(size_t)b * 2 * H_ + H_ + h] = (bf16)acc;   // ctx half of xi
    x2[(size_t)b * 2 * H_ + h]      = (bf16)er[h]; // embedding half of xi
  }
}

// ---------------------------------------------------------------------------
// GRU gate fusion: r,z,n from gi/gh, h_new = (1-z)*n + z*h
// ---------------------------------------------------------------------------
__global__ __launch_bounds__(256) void gru_update_kernel(
    const float* __restrict__ gi,   // [B,3H]
    const float* __restrict__ gh,   // [B,3H]
    const float* __restrict__ hid,  // [B,H]
    float* __restrict__ hid_new,    // [B,H]
    bf16*  __restrict__ hidb,       // [B,H] bf16 (A-matrix for next GEMMs)
    float* __restrict__ hT) {       // final hidden output slot
  const int b = blockIdx.x;
  const float* gib = gi + (size_t)b * 3 * H_;
  const float* ghb = gh + (size_t)b * 3 * H_;
  for (int h = threadIdx.x; h < H_; h += 256) {
    float ir = gib[h], iz = gib[H_ + h], inn = gib[2 * H_ + h];
    float hr = ghb[h], hz = ghb[H_ + h], hn  = ghb[2 * H_ + h];
    float r = 1.0f / (1.0f + expf(-(ir + hr)));
    float z = 1.0f / (1.0f + expf(-(iz + hz)));
    float n = tanhf(inn + r * hn);
    float hv = (1.0f - z) * n + z * hid[(size_t)b * H_ + h];
    hid_new[(size_t)b * H_ + h] = hv;
    hidb[(size_t)b * H_ + h] = (bf16)hv;
    hT[(size_t)b * H_ + h] = hv;  // last step's value survives (stream-ordered)
  }
}

// ---------------------------------------------------------------------------
// host
// ---------------------------------------------------------------------------
extern "C" void kernel_launch(void* const* d_in, const int* in_sizes, int n_in,
                              void* d_out, int out_size, void* d_ws,
                              size_t ws_size, hipStream_t stream) {
  (void)in_sizes; (void)n_in; (void)out_size; (void)ws_size;

  const float* enc    = (const float*)d_in[0];
  const float* ehid   = (const float*)d_in[1];
  const int*   target = (const int*)d_in[2];
  const float* emb    = (const float*)d_in[3];
  const float* Wa_w   = (const float*)d_in[4];
  const float* Wa_b   = (const float*)d_in[5];
  const float* Ua_w   = (const float*)d_in[6];
  const float* Ua_b   = (const float*)d_in[7];
  const float* Va_w   = (const float*)d_in[8];
  const float* Va_b   = (const float*)d_in[9];
  const float* W_ih   = (const float*)d_in[10];
  const float* W_hh   = (const float*)d_in[11];
  const float* b_ih   = (const float*)d_in[12];
  const float* b_hh   = (const float*)d_in[13];
  const float* out_w  = (const float*)d_in[14];
  const float* out_b  = (const float*)d_in[15];

  float* dec_out = (float*)d_out;                       // [B,T,V]
  float* hT      = dec_out + (size_t)B_ * T_ * V_;      // [1,B,H]
  float* attn    = hT + (size_t)B_ * H_;                // [B,T,S]

  // workspace carve
  char* ws = (char*)d_ws;
  size_t off = 0;
  auto carve = [&](size_t bytes) -> void* {
    void* p = ws + off;
    off = (off + bytes + 255) & ~(size_t)255;
    return p;
  };
  bf16*  outw16 = (bf16*)carve((size_t)V_ * H_ * 2);
  bf16*  Wih16  = (bf16*)carve((size_t)3 * H_ * 2 * H_ * 2);
  bf16*  Whh16  = (bf16*)carve((size_t)3 * H_ * H_ * 2);
  bf16*  Wa16   = (bf16*)carve((size_t)H_ * H_ * 2);
  bf16*  Ua16   = (bf16*)carve((size_t)H_ * H_ * 2);
  bf16*  enc16  = (bf16*)carve((size_t)B_ * S_ * H_ * 2);
  float* Uk     = (float*)carve((size_t)B_ * S_ * H_ * 4);
  float* hidf0  = (float*)carve((size_t)B_ * H_ * 4);
  float* hidf1  = (float*)carve((size_t)B_ * H_ * 4);
  bf16*  hidb   = (bf16*)carve((size_t)B_ * H_ * 2);
  float* qbuf   = (float*)carve((size_t)B_ * H_ * 4);
  float* ctxbuf = (float*)carve((size_t)B_ * H_ * 4);
  bf16*  x2buf  = (bf16*)carve((size_t)B_ * 2 * H_ * 2);
  float* gibuf  = (float*)carve((size_t)B_ * 3 * H_ * 4);
  float* ghbuf  = (float*)carve((size_t)B_ * 3 * H_ * 4);

  auto cvt = [&](const float* s, bf16* d, long n) {
    f32_to_bf16_kernel<<<(unsigned)((n + 255) / 256), 256, 0, stream>>>(s, d, n);
  };
  // one-time (per launch) weight conversion to bf16 — resident in 192MB L2
  cvt(out_w, outw16, (long)V_ * H_);
  cvt(W_ih,  Wih16,  (long)3 * H_ * 2 * H_);
  cvt(W_hh,  Whh16,  (long)3 * H_ * H_);
  cvt(Wa_w,  Wa16,   (long)H_ * H_);
  cvt(Ua_w,  Ua16,   (long)H_ * H_);
  cvt(enc,   enc16,  (long)B_ * S_ * H_);
  init_hidden_kernel<<<(B_ * H_ + 255) / 256, 256, 0, stream>>>(ehid, hidf0, hidb);

  const size_t shA1K = (size_t)32 * H_ * 2;      // 64 KB LDS (K=1024)
  const size_t shA2K = (size_t)32 * 2 * H_ * 2;  // 128 KB LDS (K=2048)

  // Uk = enc @ Ua.T + Ua_b : M=2048 (64 m-tiles), N=1024, K=1024
  {
    dim3 g(H_ / 128, (B_ * S_) / 32);
    gemm_bf16_wmma_kernel<<<g, 256, shA1K, stream>>>(enc16, Ua16, Ua_b, Uk,
                                                     H_, (long)H_, 0);
  }

  float* hcur = hidf0;
  float* hnext = hidf1;
  for (int t = 0; t < T_; ++t) {
    // q = hid @ Wa.T + Wa_b : 32 x 1024, K=1024
    {
      dim3 g(H_ / 128, 1);
      gemm_bf16_wmma_kernel<<<g, 256, shA1K, stream>>>(hidb, Wa16, Wa_b, qbuf,
                                                       H_, (long)H_, 0);
    }
    // attention + context + build xi=[emb[tok], ctx] (bf16)
    attn_ctx_kernel<<<B_, 256, 0, stream>>>(qbuf, Uk, Va_w, Va_b, enc, target,
                                            emb, ctxbuf, x2buf, attn, t);
    // fused gate GEMMs: z=0: gi (K=2048), z=1: gh (K=1024); N=3072 each
    {
      dim3 g((3 * H_) / 128, 1, 2);
      gemm_gates_kernel<<<g, 256, shA2K, stream>>>(
          x2buf, Wih16, b_ih, gibuf, 2 * H_,
          hidb,  Whh16, b_hh, ghbuf, H_, (long)(3 * H_));
    }
    // GRU gating -> h_new (f32 + bf16)
    gru_update_kernel<<<B_, 256, 0, stream>>>(gibuf, ghbuf, hcur, hnext, hidb,
                                              hT);
    // logits = h_new @ out_w.T + out_b : 32 x 32000, K=1024 (dominant GEMM)
    // nontemporal stores: 262MB streaming output must not evict L2 weights
    {
      dim3 g(V_ / 128, 1);
      gemm_bf16_wmma_kernel<<<g, 256, shA1K, stream>>>(
          hidb, outw16, out_b, dec_out + (size_t)t * V_, H_,
          (long)T_ * V_, 1);
    }
    float* tmp = hcur; hcur = hnext; hnext = tmp;
  }
}
